// VQGAN_82660940578848
// MI455X (gfx1250) — matrix operand order repaired and checked by reading, not compiled, not run on hardware
//
#include <hip/hip_runtime.h>
#include <math.h>

#define BATCH 2
#define NGROUPS 32

typedef __attribute__((ext_vector_type(16))) __bf16 v16bf;
typedef __attribute__((ext_vector_type(16))) unsigned short v16u;
typedef __attribute__((ext_vector_type(8))) float v8f;

static __device__ __forceinline__ unsigned short f2bfu(float f) {
  unsigned u = __builtin_bit_cast(unsigned, f);
  return (unsigned short)((u + 0x7FFFu + ((u >> 16) & 1u)) >> 16);
}

// ---------------------------------------------------------------------------
// Implicit-GEMM conv via v_wmma_f32_16x16x32_bf16.
// GEMM view: M = Cout, N = B*Hout*Wout, K = Cin*KS*KS.
// One 16x16 output tile per wave32.
// A fragment (16x32 bf16): lane&15 = row M, lane>>4 selects K subrange.
//   elem i -> K = half*16 + i
// B fragment (32x16 bf16): lane&15 = col N; lane<16 covers K=0..15, lane>=16 K=16..31.
// C/D (f32, 8 VGPR): lane&15 = col N; VGPR r -> M = half*8 + r.
// ---------------------------------------------------------------------------
template <int KS>
__global__ void __launch_bounds__(256) conv_wmma_kernel(
    const float* __restrict__ in, const float* __restrict__ wt,
    const float* __restrict__ bias, const float* __restrict__ resid,
    float* __restrict__ out,
    int Cin, int Cout, int Hin, int Win, int Hout, int Wout,
    int stride, int pad)
{
  const int HWo   = Hout * Wout;
  const int Npix  = BATCH * HWo;
  const int Ktot  = Cin * KS * KS;
  const int tilesM = (Cout + 15) >> 4;
  const int tilesN = (Npix + 15) >> 4;
  int wave = blockIdx.x * (blockDim.x >> 5) + (threadIdx.x >> 5);
  if (wave >= tilesM * tilesN) return;
  int tm = wave % tilesM, tn = wave / tilesM;
  int m0 = tm << 4, n0 = tn << 4;
  int lane = threadIdx.x & 31;
  int half = lane >> 4, l15 = lane & 15;
  int mrow = m0 + l15;
  int ncol = n0 + l15;
  bool mv = mrow < Cout;
  bool nv = ncol < Npix;
  int pb = 0, py = 0, px = 0;
  if (nv) { pb = ncol / HWo; int p = ncol - pb * HWo; py = p / Wout; px = p - py * Wout; }
  const float* wrow = wt + (long)mrow * Ktot;
  v8f acc = {};
  for (int k0 = 0; k0 < Ktot; k0 += 32) {
    v16u au, bu;
#pragma unroll
    for (int i = 0; i < 16; ++i) {
      int ka = k0 + half * 16 + i;
      float av = (mv && ka < Ktot) ? wrow[ka] : 0.f;
      au[i] = f2bfu(av);
      int kb = k0 + (half ? 16 + i : i);
      float bv = 0.f;
      if (nv && kb < Ktot) {
        int ci = kb / (KS * KS);
        int r  = kb - ci * (KS * KS);
        int ky = r / KS, kx = r - ky * KS;
        int yi = py * stride + ky - pad;
        int xi = px * stride + kx - pad;
        if (yi >= 0 && yi < Hin && xi >= 0 && xi < Win)
          bv = in[(((long)pb * Cin + ci) * Hin + yi) * Win + xi];
      }
      bu[i] = f2bfu(bv);
    }
    v16bf a = __builtin_bit_cast(v16bf, au);
    v16bf b = __builtin_bit_cast(v16bf, bu);
    acc = __builtin_amdgcn_wmma_f32_16x16x32_bf16(false, a, false, b, (short)0, acc,
                                                  false, false);
  }
  if (!nv) return;
  long outPix = (long)pb * Cout * HWo + (long)py * Wout + px;
#pragma unroll
  for (int r = 0; r < 8; ++r) {
    int m = m0 + half * 8 + r;
    if (m < Cout) {
      long oi = outPix + (long)m * HWo;
      float v = acc[r] + bias[m];
      if (resid) v += resid[oi];
      out[oi] = v;
    }
  }
}

// ---------------------------------------------------------------------------
// Generic batched fp32->bf16 WMMA GEMM: C[m,n] = sum_k A(m,k)*B(k,n)
// A(m,k) = transA ? A[k*lda+m] : A[m*lda+k];  B(k,n) = transB ? B[n*ldb+k] : B[k*ldb+n]
// ---------------------------------------------------------------------------
__global__ void __launch_bounds__(256) gemm_wmma_kernel(
    const float* __restrict__ A, const float* __restrict__ B, float* __restrict__ C,
    int M, int N, int K, int lda, int ldb, int ldc, int transA, int transB,
    long sA, long sB, long sC)
{
  A += (long)blockIdx.y * sA; B += (long)blockIdx.y * sB; C += (long)blockIdx.y * sC;
  int tilesM = (M + 15) >> 4, tilesN = (N + 15) >> 4;
  int wave = blockIdx.x * (blockDim.x >> 5) + (threadIdx.x >> 5);
  if (wave >= tilesM * tilesN) return;
  int tm = wave % tilesM, tn = wave / tilesM;
  int m0 = tm << 4, n0 = tn << 4;
  int lane = threadIdx.x & 31;
  int half = lane >> 4, l15 = lane & 15;
  int mrow = m0 + l15, ncol = n0 + l15;
  bool mv = mrow < M, nv = ncol < N;
  v8f acc = {};
  for (int k0 = 0; k0 < K; k0 += 32) {
    v16u au, bu;
#pragma unroll
    for (int i = 0; i < 16; ++i) {
      int ka = k0 + half * 16 + i;
      float av = 0.f;
      if (mv && ka < K) av = transA ? A[(long)ka * lda + mrow] : A[(long)mrow * lda + ka];
      au[i] = f2bfu(av);
      int kb = k0 + (half ? 16 + i : i);
      float bv = 0.f;
      if (nv && kb < K) bv = transB ? B[(long)ncol * ldb + kb] : B[(long)kb * ldb + ncol];
      bu[i] = f2bfu(bv);
    }
    v16bf a = __builtin_bit_cast(v16bf, au);
    v16bf b = __builtin_bit_cast(v16bf, bu);
    acc = __builtin_amdgcn_wmma_f32_16x16x32_bf16(false, a, false, b, (short)0, acc,
                                                  false, false);
  }
  if (!nv) return;
#pragma unroll
  for (int r = 0; r < 8; ++r) {
    int m = m0 + half * 8 + r;
    if (m < M) C[(long)m * ldc + ncol] = acc[r];
  }
}

// ---------------------------------------------------------------------------
// GroupNorm (+ optional swish). One block per (batch, group).
// ---------------------------------------------------------------------------
__global__ void __launch_bounds__(256) groupnorm_kernel(
    const float* __restrict__ x, const float* __restrict__ gam,
    const float* __restrict__ bet, float* __restrict__ y,
    int C, int HW, int doSwish)
{
  int cpg = C / NGROUPS;
  int b = blockIdx.x / NGROUPS, gr = blockIdx.x % NGROUPS;
  const float* xp = x + ((long)b * C + (long)gr * cpg) * HW;
  float* yp = y + ((long)b * C + (long)gr * cpg) * HW;
  long n = (long)cpg * HW;
  float s = 0.f, ss = 0.f;
  for (long i = threadIdx.x; i < n; i += blockDim.x) { float v = xp[i]; s += v; ss += v * v; }
  __shared__ float rs[256], rq[256];
  rs[threadIdx.x] = s; rq[threadIdx.x] = ss;
  __syncthreads();
  for (int o = 128; o > 0; o >>= 1) {
    if ((int)threadIdx.x < o) { rs[threadIdx.x] += rs[threadIdx.x + o]; rq[threadIdx.x] += rq[threadIdx.x + o]; }
    __syncthreads();
  }
  float mean = rs[0] / (float)n;
  float var  = rq[0] / (float)n - mean * mean;
  float inv  = rsqrtf(var + 1e-6f);
  for (long i = threadIdx.x; i < n; i += blockDim.x) {
    int c = gr * cpg + (int)(i / HW);
    float v = (xp[i] - mean) * inv * gam[c] + bet[c];
    if (doSwish) v = v / (1.f + expf(-v));
    yp[i] = v;
  }
}

// Row softmax of softmax(scale * g). One block per row.
__global__ void __launch_bounds__(256) softmax_kernel(float* __restrict__ G, int cols, float scale)
{
  float* g = G + (long)blockIdx.x * cols;
  __shared__ float red[256];
  float mx = -3.4e38f;
  for (int i = threadIdx.x; i < cols; i += blockDim.x) mx = fmaxf(mx, g[i]);
  red[threadIdx.x] = mx; __syncthreads();
  for (int o = 128; o > 0; o >>= 1) {
    if ((int)threadIdx.x < o) red[threadIdx.x] = fmaxf(red[threadIdx.x], red[threadIdx.x + o]);
    __syncthreads();
  }
  mx = red[0]; __syncthreads();
  float s = 0.f;
  for (int i = threadIdx.x; i < cols; i += blockDim.x) {
    float e = expf(scale * (g[i] - mx)); g[i] = e; s += e;
  }
  red[threadIdx.x] = s; __syncthreads();
  for (int o = 128; o > 0; o >>= 1) {
    if ((int)threadIdx.x < o) red[threadIdx.x] += red[threadIdx.x + o];
    __syncthreads();
  }
  float inv = 1.f / red[0];
  for (int i = threadIdx.x; i < cols; i += blockDim.x) g[i] *= inv;
}

__global__ void __launch_bounds__(256) upsample2_kernel(
    const float* __restrict__ in, float* __restrict__ out, int C, int Hin, int Win)
{
  long t = (long)blockIdx.x * blockDim.x + threadIdx.x;
  int Ho = Hin * 2, Wo = Win * 2;
  long total = (long)BATCH * C * Ho * Wo;
  if (t >= total) return;
  int x = (int)(t % Wo); long r = t / Wo;
  int y = (int)(r % Ho); r /= Ho;
  int c = (int)(r % C);  int b = (int)(r / C);
  out[t] = in[(((long)b * C + c) * Hin + (y >> 1)) * Win + (x >> 1)];
}

__global__ void __launch_bounds__(256) enorm_kernel(
    const float* __restrict__ embed, float* __restrict__ en, int Kc, int D)
{
  int k = blockIdx.x * blockDim.x + threadIdx.x;
  if (k >= Kc) return;
  float s = 0.f;
  for (int d = 0; d < D; ++d) { float v = embed[(long)k * D + d]; s += v * v; }
  en[k] = s;
}

// one block per pixel row of G; d(k) = |e_k|^2 - 2 z.e_k  (|z|^2 constant per row)
__global__ void __launch_bounds__(256) argmin_kernel(
    const float* __restrict__ G, const float* __restrict__ enorm,
    int* __restrict__ idx, float* __restrict__ counts, int Kc)
{
  const float* gr = G + (long)blockIdx.x * Kc;
  float best = 3.4e38f; int bi = 0;
  for (int k = threadIdx.x; k < Kc; k += blockDim.x) {
    float d = enorm[k] - 2.f * gr[k];
    if (d < best) { best = d; bi = k; }
  }
  __shared__ float bv[256]; __shared__ int bx[256];
  bv[threadIdx.x] = best; bx[threadIdx.x] = bi; __syncthreads();
  for (int o = 128; o > 0; o >>= 1) {
    if ((int)threadIdx.x < o) {
      float ov = bv[threadIdx.x + o]; int oi = bx[threadIdx.x + o];
      if (ov < bv[threadIdx.x] || (ov == bv[threadIdx.x] && oi < bx[threadIdx.x])) {
        bv[threadIdx.x] = ov; bx[threadIdx.x] = oi;
      }
    }
    __syncthreads();
  }
  if (threadIdx.x == 0) { idx[blockIdx.x] = bx[0]; atomicAdd(&counts[bx[0]], 1.f); }
}

__global__ void __launch_bounds__(256) gather_loss_kernel(
    const float* __restrict__ z, const float* __restrict__ embed,
    const int* __restrict__ idx, float* __restrict__ zq, float* __restrict__ sumsq,
    int D, int HW)
{
  int t = blockIdx.x * blockDim.x + threadIdx.x;
  int total = BATCH * HW * D;
  float dsq = 0.f;
  if (t < total) {
    int d = t % D, n = t / D;
    int b = n / HW, p = n % HW;
    long zo = ((long)b * D + d) * HW + p;
    float zv = z[zo];
    float qv = embed[(long)idx[n] * D + d];
    zq[zo] = qv;
    float df = qv - zv; dsq = df * df;
  }
  __shared__ float red[256];
  red[threadIdx.x] = dsq; __syncthreads();
  for (int o = 128; o > 0; o >>= 1) {
    if ((int)threadIdx.x < o) red[threadIdx.x] += red[threadIdx.x + o];
    __syncthreads();
  }
  if (threadIdx.x == 0) atomicAdd(sumsq, red[0]);
}

__global__ void __launch_bounds__(256) finalize_kernel(
    const float* __restrict__ sumsq, const float* __restrict__ counts,
    float* __restrict__ outLP, int Kc, float Npix, float totalElems)
{
  __shared__ float red[256];
  float s = 0.f;
  for (int k = threadIdx.x; k < Kc; k += blockDim.x) {
    float em = counts[k] / Npix;
    s += em * logf(em + 1e-10f);
  }
  red[threadIdx.x] = s; __syncthreads();
  for (int o = 128; o > 0; o >>= 1) {
    if ((int)threadIdx.x < o) red[threadIdx.x] += red[threadIdx.x + o];
    __syncthreads();
  }
  if (threadIdx.x == 0) {
    outLP[0] = 1.25f * sumsq[0] / totalElems;  // (1+BETA)*mse
    outLP[1] = expf(-red[0]);
  }
}

// ===========================================================================
// Host-side orchestration
// ===========================================================================
enum { K_CONV3, K_RES, K_ATTN, K_DOWN, K_UP, K_NORM };
struct Layer { int kind, cin, cout; };

static const Layer ENC[16] = {
  {K_CONV3,3,64},{K_RES,64,64},{K_RES,64,64},{K_DOWN,64,64},
  {K_RES,64,128},{K_RES,128,128},{K_DOWN,128,128},
  {K_RES,128,256},{K_ATTN,256,256},{K_RES,256,256},{K_ATTN,256,256},
  {K_RES,256,256},{K_ATTN,256,256},{K_RES,256,256},{K_NORM,256,256},{K_CONV3,256,256}};

static const Layer GEN[16] = {
  {K_CONV3,256,256},{K_RES,256,256},{K_ATTN,256,256},{K_RES,256,256},
  {K_RES,256,256},{K_ATTN,256,256},{K_RES,256,256},{K_ATTN,256,256},{K_UP,256,256},
  {K_RES,256,128},{K_RES,128,128},{K_UP,128,128},
  {K_RES,128,64},{K_RES,64,64},{K_NORM,64,64},{K_CONV3,64,3}};

struct ConvP { const float *w, *b; };
struct NormP { const float *g, *b; };
struct LayerP {
  ConvP c;  NormP n;
  NormP n1, n2; ConvP c1, c2, co; int has_co;
  NormP an; ConvP q, k, v, proj;
};

extern "C" void kernel_launch(void* const* d_in, const int* in_sizes, int n_in,
                              void* d_out, int out_size, void* d_ws, size_t ws_size,
                              hipStream_t stream) {
  (void)out_size; (void)ws_size;
  // Input flattening order: either jax-pytree sorted keys (embed first, x last)
  // or python insertion order (x first, embed last). Distinguish via in_sizes[0].
  const bool sortedMode = (in_sizes[0] == 1024 * 256);
  const float* embed; const float* xin;
  int cur = 1;
  if (sortedMode) { embed = (const float*)d_in[0]; xin = (const float*)d_in[n_in - 1]; }
  else            { xin   = (const float*)d_in[0]; embed = (const float*)d_in[n_in - 1]; }

  auto nf = [&]() { return (const float*)d_in[cur++]; };
  auto rdConv = [&](ConvP& c) { if (sortedMode) { c.b = nf(); c.w = nf(); } else { c.w = nf(); c.b = nf(); } };
  auto rdNorm = [&](NormP& n) { if (sortedMode) { n.b = nf(); n.g = nf(); } else { n.g = nf(); n.b = nf(); } };

  LayerP encP[16], genP[16];
  auto parse = [&](const Layer* L, LayerP* P) {
    for (int i = 0; i < 16; ++i) {
      LayerP& p = P[i];
      switch (L[i].kind) {
        case K_CONV3: case K_DOWN: case K_UP: rdConv(p.c); break;
        case K_NORM: rdNorm(p.n); break;
        case K_RES:
          p.has_co = (L[i].cin != L[i].cout);
          if (sortedMode) { rdConv(p.c1); rdConv(p.c2); if (p.has_co) rdConv(p.co); rdNorm(p.n1); rdNorm(p.n2); }
          else            { rdNorm(p.n1); rdConv(p.c1); rdNorm(p.n2); rdConv(p.c2); if (p.has_co) rdConv(p.co); }
          break;
        case K_ATTN:
          if (sortedMode) { rdConv(p.k); rdNorm(p.an); rdConv(p.proj); rdConv(p.q); rdConv(p.v); }
          else            { rdNorm(p.an); rdConv(p.q); rdConv(p.k); rdConv(p.v); rdConv(p.proj); }
          break;
      }
    }
  };
  parse(ENC, encP);
  parse(GEN, genP);

  // ----- workspace carve-out -----
  float* Wf = (float*)d_ws;
  size_t off = 0;
  auto wsAlloc = [&](size_t nfl) { float* p = Wf + off; off += nfl; return p; };
  float* B4[4]; for (int i = 0; i < 4; ++i) B4[i] = wsAlloc((size_t)4 * 1024 * 1024);
  float* attq = wsAlloc(524288); float* attk = wsAlloc(524288);
  float* attv = wsAlloc(524288); float* atth = wsAlloc(524288);
  float* Gbuf  = wsAlloc((size_t)2 * 1024 * 1024);
  float* enorm = wsAlloc(1024);
  float* counts = wsAlloc(1024);
  float* sumsq  = wsAlloc(16);
  int*   idxb   = (int*)wsAlloc(2048);

  // ----- launch helpers -----
  auto convL = [&](const float* in, float* out, const ConvP& p, int Cin, int Cout,
                   int Hin, int Hout, int ks, int stride, int pad, const float* res) {
    int Npix = BATCH * Hout * Hout;
    int tiles = ((Cout + 15) / 16) * ((Npix + 15) / 16);
    int blocks = (tiles + 7) / 8;
    if (ks == 3)
      conv_wmma_kernel<3><<<blocks, 256, 0, stream>>>(in, p.w, p.b, res, out,
          Cin, Cout, Hin, Hin, Hout, Hout, stride, pad);
    else
      conv_wmma_kernel<1><<<blocks, 256, 0, stream>>>(in, p.w, p.b, res, out,
          Cin, Cout, Hin, Hin, Hout, Hout, stride, pad);
  };
  auto gnL = [&](const float* in, float* out, const NormP& p, int C, int H, int doSwish) {
    groupnorm_kernel<<<BATCH * NGROUPS, 256, 0, stream>>>(in, p.g, p.b, out, C, H * H, doSwish);
  };
  auto gemmL = [&](const float* A, const float* Bm, float* C, int M, int N, int K,
                   int lda, int ldb, int ldc, int tA, int tB, long sA, long sB, long sC, int batch) {
    int tiles = ((M + 15) / 16) * ((N + 15) / 16);
    dim3 g((tiles + 7) / 8, batch);
    gemm_wmma_kernel<<<g, 256, 0, stream>>>(A, Bm, C, M, N, K, lda, ldb, ldc, tA, tB, sA, sB, sC);
  };
  auto pick = [](int a, int b, int c) { for (int i = 0; i < 4; ++i) if (i != a && i != b && i != c) return i; return 0; };

  struct Cur { const float* p; int idx; };
  auto runPlan = [&](const Layer* L, const LayerP* P, const float* input, int inIdx,
                     int H0, float* lastOut) -> Cur {
    const float* curp = input;
    int ci = inIdx;
    int H = H0;
    for (int li = 0; li < 16; ++li) {
      const Layer& ly = L[li];
      const LayerP& p = P[li];
      bool isLast = (li == 15);
      if (ly.kind == K_CONV3) {
        int o = pick(ci, -1, -1);
        float* out = (isLast && lastOut) ? lastOut : B4[o];
        convL(curp, out, p.c, ly.cin, ly.cout, H, H, 3, 1, 1, nullptr);
        curp = out; ci = (isLast && lastOut) ? -1 : o;
      } else if (ly.kind == K_DOWN) {
        // reference: pad (0,1)x(0,1) then stride-2 pad-0 conv == OOB-high reads as zero
        int o = pick(ci, -1, -1);
        convL(curp, B4[o], p.c, ly.cin, ly.cout, H, H / 2, 3, 2, 0, nullptr);
        H /= 2; curp = B4[o]; ci = o;
      } else if (ly.kind == K_UP) {
        int t = pick(ci, -1, -1);
        long total = (long)BATCH * ly.cin * (2 * H) * (2 * H);
        upsample2_kernel<<<(int)((total + 255) / 256), 256, 0, stream>>>(curp, B4[t], ly.cin, H, H);
        H *= 2;
        int o = pick(ci, t, -1);
        convL(B4[t], B4[o], p.c, ly.cin, ly.cout, H, H, 3, 1, 1, nullptr);
        curp = B4[o]; ci = o;
      } else if (ly.kind == K_NORM) {
        int o = pick(ci, -1, -1);
        gnL(curp, B4[o], p.n, ly.cout, H, 0);
        curp = B4[o]; ci = o;
      } else if (ly.kind == K_RES) {
        int t1 = pick(ci, -1, -1);
        gnL(curp, B4[t1], p.n1, ly.cin, H, 1);                         // swish(gn(x))
        int t2 = pick(ci, t1, -1);
        convL(B4[t1], B4[t2], p.c1, ly.cin, ly.cout, H, H, 3, 1, 1, nullptr);
        gnL(B4[t2], B4[t1], p.n2, ly.cout, H, 1);
        const float* res = curp;
        if (p.has_co) {
          int t3 = pick(ci, t1, t2);
          convL(curp, B4[t3], p.co, ly.cin, ly.cout, H, H, 1, 1, 0, nullptr);
          res = B4[t3];
        }
        convL(B4[t1], B4[t2], p.c2, ly.cout, ly.cout, H, H, 3, 1, 1, res);
        curp = B4[t2]; ci = t2;
      } else { // K_ATTN (H=32, C=256)
        int C = ly.cin, HW = H * H;
        int t1 = pick(ci, -1, -1);
        gnL(curp, B4[t1], p.an, C, H, 0);
        convL(B4[t1], attq, p.q, C, C, H, H, 1, 1, 0, nullptr);
        convL(B4[t1], attk, p.k, C, C, H, H, 1, 1, 0, nullptr);
        convL(B4[t1], attv, p.v, C, C, H, H, 1, 1, 0, nullptr);
        // G[b,qp,kp] = sum_c q[b,c,qp]*k[b,c,kp]
        gemmL(attq, attk, Gbuf, HW, HW, C, HW, HW, HW, 1, 0,
              (long)C * HW, (long)C * HW, (long)HW * HW, BATCH);
        float scale = 1.0f / sqrtf((float)C);
        softmax_kernel<<<BATCH * HW, 256, 0, stream>>>(Gbuf, HW, scale);
        // h[b,c,qp] = sum_kp v[b,c,kp]*G[b,qp,kp]
        gemmL(attv, Gbuf, atth, C, HW, HW, HW, HW, HW, 0, 1,
              (long)C * HW, (long)HW * HW, (long)C * HW, BATCH);
        int o = pick(ci, t1, -1);
        convL(atth, B4[o], p.proj, C, C, H, H, 1, 1, 0, curp);  // x + proj(h)
        curp = B4[o]; ci = o;
      }
    }
    return {curp, ci};
  };

  // -------- encoder --------
  Cur z = runPlan(ENC, encP, xin, -1, 128, nullptr);

  // -------- quantize --------
  enorm_kernel<<<4, 256, 0, stream>>>(embed, enorm, 1024, 256);
  hipMemsetAsync(counts, 0, (1024 + 16) * sizeof(float), stream);
  // G[b*1024 + p, k] = z[b,:,p] . embed[k,:]
  gemmL(z.p, embed, Gbuf, 1024, 1024, 256, 1024, 256, 1024, 1, 1,
        (long)256 * 1024, 0, (long)1024 * 1024, BATCH);
  argmin_kernel<<<2048, 256, 0, stream>>>(Gbuf, enorm, idxb, counts, 1024);
  int zqIdx = pick(z.idx, -1, -1);
  float* zq = B4[zqIdx];
  gather_loss_kernel<<<2048, 256, 0, stream>>>(z.p, embed, idxb, zq, sumsq, 256, 1024);

  // -------- generator --------
  float* outf = (float*)d_out;
  runPlan(GEN, genP, zq, zqIdx, 32, outf);

  finalize_kernel<<<1, 256, 0, stream>>>(sumsq, counts, outf + BATCH * 3 * 128 * 128,
                                         1024, 2048.f, 524288.f);
}
